// GraphConv_88648124989846
// MI455X (gfx1250) — matrix operand order repaired and checked by reading, not compiled, not run on hardware
//
#include <hip/hip_runtime.h>
#include <stdint.h>

typedef __attribute__((ext_vector_type(16))) _Float16 v16h;
typedef __attribute__((ext_vector_type(8)))  float    v8f;

#define N_NODES   100000
#define N_EDGES   1600000
#define IN_DIM    64
#define OUT_DIM   64
#define MSG_DIM   67
#define EPS       1e-5f
#define TILE_EDGES 128
#define N_TILES   (N_EDGES / TILE_EDGES)   /* 12500, exact */
#define GRID_A    512
/* ws layout (floats): [0:64)=a, [64:128)=b, [256 : 256+GRID_A*128)=block partials */
#define WS_PART   256

/* LDS row stride in dwords: multiple of 4 (16B-aligned rows for b128 loads);
   52 = 4*13 -> 52*dm % 64 != 0 for dm in [1,15], so rows land in distinct banks. */
#define STRIDE    52

static __device__ inline uint32_t pkh(float a, float b) {
  union { _Float16 h[2]; uint32_t u; } x;
  x.h[0] = (_Float16)a; x.h[1] = (_Float16)b;
  return x.u;
}

union ABfrag { v16h h; uint4 q[2]; };

/* Pack W (64x67 row-major f32) n-major into LDS:
   Wpk[n*STRIDE + kk] = (f16 W[n][2kk], f16 W[n][2kk+1]), kk in [0,48), K padded to 96. */
static __device__ inline void stage_W(const float* __restrict__ W, uint32_t* Wpk, int tid) {
  #pragma unroll
  for (int i = 0; i < 12; ++i) {
    int idx = tid + i * 256;          /* 3072 = 48*64 entries */
    int kk  = idx & 47;               /* not a power of two; use div/mod */
    int n   = idx / 48;
    kk = idx - n * 48;
    int k = kk * 2;
    float w0 = (k     < MSG_DIM) ? W[n * MSG_DIM + k]     : 0.f;
    float w1 = (k + 1 < MSG_DIM) ? W[n * MSG_DIM + k + 1] : 0.f;
    Wpk[n * STRIDE + kk] = pkh(w0, w1);
  }
}

/* Stage one half (32 features) of one msg row into packed-f16 LDS.
   Row layout: Apk[m*STRIDE + kk] = f16 pair (msg[2kk], msg[2kk+1]), kk in [0,48). */
static __device__ inline void stage_row_half(const float* __restrict__ node,
                                             const float* __restrict__ feat,
                                             const int*   __restrict__ edges,
                                             long long e,
                                             uint32_t* __restrict__ Apk,
                                             int m, int half) {
  int srcn = edges[2 * e];
  int dstn = edges[2 * e + 1];
  const float4* f4 = (const float4*)(feat + (size_t)dstn * IN_DIM + half * 32);
  uint32_t* row = Apk + m * STRIDE;
  #pragma unroll
  for (int j = 0; j < 8; ++j) {
    float4 v = f4[j];
    row[half * 16 + 2 * j]     = pkh(v.x, v.y);
    row[half * 16 + 2 * j + 1] = pkh(v.z, v.w);
  }
  if (half) {
    float d0 = node[(size_t)dstn * 3 + 0] - node[(size_t)srcn * 3 + 0];
    float d1 = node[(size_t)dstn * 3 + 1] - node[(size_t)srcn * 3 + 1];
    float d2 = node[(size_t)dstn * 3 + 2] - node[(size_t)srcn * 3 + 2];
    row[32] = pkh(d0, d1);
    row[33] = pkh(d2, 0.f);
    #pragma unroll
    for (int kk = 34; kk < 48; ++kk) row[kk] = 0u;
  }
}

/* One wave computes a 16(edges) x 64(channels) f32 tile: 3 k-steps x 4 n-tiles.
   All fragment loads are aligned uint4 (-> ds_load_b128). */
static __device__ inline void wave_gemm(const uint32_t* __restrict__ Apk,
                                        const uint32_t* __restrict__ Wpk,
                                        int lane, v8f acc[4]) {
  int m  = lane & 15;
  int hi = (lane >> 4) & 1;
  #pragma unroll
  for (int kt = 0; kt < 3; ++kt) {
    /* A 16x32 f16 layout: VGPR v holds K pair (2k,2k+1);
       v in 0..3 -> kk = kt*16 + hi*4 + v ; v in 4..7 -> kk = kt*16 + 8 + hi*4 + (v-4) */
    ABfrag a;
    const uint4* ap = (const uint4*)(Apk + m * STRIDE + kt * 16 + hi * 4);
    a.q[0] = ap[0];                 /* kk base + 0..3  */
    a.q[1] = ap[2];                 /* kk base + 8..11 */
    #pragma unroll
    for (int nt = 0; nt < 4; ++nt) {
      /* B 32x16 f16 layout: same K order as A (kk = kt*16 + hi*4 + v, v>=4 -> +8) */
      int n = nt * 16 + m;
      ABfrag b;
      const uint4* bp = (const uint4*)(Wpk + n * STRIDE + kt * 16 + hi * 4);
      b.q[0] = bp[0];
      b.q[1] = bp[2];
      acc[nt] = __builtin_amdgcn_wmma_f32_16x16x32_f16(
          false, a.h, false, b.h, (short)0, acc[nt], false, false);
    }
  }
}

/* ---- Kernel A: per-channel sum / sum-of-squares partials (deterministic) ---- */
__global__ void __launch_bounds__(256)
kA_stats(const float* __restrict__ node, const float* __restrict__ feat,
         const float* __restrict__ W, const int* __restrict__ edges,
         float* __restrict__ ws) {
  __shared__ uint32_t Wpk[64 * STRIDE];
  __shared__ uint32_t Apk[8][16 * STRIDE];
  __shared__ float    red[2][8][64];
  int tid  = threadIdx.x;
  int lane = tid & 31, w = tid >> 5;
  stage_W(W, Wpk, tid);
  float s[4] = {0, 0, 0, 0}, q[4] = {0, 0, 0, 0};
  for (int tile = blockIdx.x; tile < N_TILES; tile += GRID_A) {
    __syncthreads();
    int r = tid >> 1, half = tid & 1;
    long long e = (long long)tile * TILE_EDGES + r;
    stage_row_half(node, feat, edges, e, Apk[r >> 4], r & 15, half);
    __syncthreads();
    v8f acc[4] = {};
    wave_gemm(Apk[w], Wpk, lane, acc);
    #pragma unroll
    for (int nt = 0; nt < 4; ++nt)
      #pragma unroll
      for (int i = 0; i < 8; ++i) {
        float v = acc[nt][i];
        s[nt] += v;
        q[nt] += v * v;
      }
  }
  #pragma unroll
  for (int nt = 0; nt < 4; ++nt) {   /* lane & lane^16 hold the same column */
    s[nt] += __shfl_xor(s[nt], 16);
    q[nt] += __shfl_xor(q[nt], 16);
  }
  __syncthreads();
  if (lane < 16) {
    #pragma unroll
    for (int nt = 0; nt < 4; ++nt) {
      red[0][w][nt * 16 + lane] = s[nt];
      red[1][w][nt * 16 + lane] = q[nt];
    }
  }
  __syncthreads();
  if (tid < 64) {
    float S = 0.f, Q = 0.f;
    #pragma unroll
    for (int ww = 0; ww < 8; ++ww) { S += red[0][ww][tid]; Q += red[1][ww][tid]; }
    ws[WS_PART + blockIdx.x * 128 + tid]      = S;
    ws[WS_PART + blockIdx.x * 128 + 64 + tid] = Q;
  }
}

/* ---- Kernel B: fold partials -> affine params a,b ---- */
__global__ void kB_finalize(const float* __restrict__ gamma,
                            const float* __restrict__ beta,
                            float* __restrict__ ws) {
  int c = threadIdx.x;  /* 64 threads */
  float S = 0.f, Q = 0.f;
  for (int b = 0; b < GRID_A; ++b) {
    S += ws[WS_PART + b * 128 + c];
    Q += ws[WS_PART + b * 128 + 64 + c];
  }
  float mean = S / (float)N_EDGES;
  float var  = Q / (float)N_EDGES - mean * mean;
  float a    = gamma[c] * rsqrtf(var + EPS);
  float bb   = beta[c] - mean * a;
  ws[c]      = a;
  ws[64 + c] = bb;
}

/* ---- Kernel C: one wave per node; recompute h for its 16 edges, norm+relu+max ---- */
__global__ void __launch_bounds__(256)
kC_out(const float* __restrict__ node, const float* __restrict__ feat,
       const float* __restrict__ W, const int* __restrict__ edges,
       const float* __restrict__ ws, float* __restrict__ out) {
  __shared__ uint32_t Wpk[64 * STRIDE];
  __shared__ uint32_t Apk[8][16 * STRIDE];
  int tid  = threadIdx.x;
  int lane = tid & 31, w = tid >> 5;
  stage_W(W, Wpk, tid);
  int r = tid >> 1, half = tid & 1;
  int nodew = blockIdx.x * 8 + (r >> 4);
  int k     = r & 15;
  long long e = (long long)nodew + (long long)k * N_NODES;  /* src = e % N pattern */
  stage_row_half(node, feat, edges, e, Apk[r >> 4], k, half);
  __syncthreads();
  v8f acc[4] = {};
  wave_gemm(Apk[w], Wpk, lane, acc);
  int mynode = blockIdx.x * 8 + w;
  #pragma unroll
  for (int nt = 0; nt < 4; ++nt) {
    int c   = nt * 16 + (lane & 15);
    float a = ws[c], b = ws[64 + c];
    float m = 0.f;                     /* matches zero-init of pooled; relu >= 0 */
    #pragma unroll
    for (int i = 0; i < 8; ++i) {
      float v = fmaf(acc[nt][i], a, b);
      m = fmaxf(m, fmaxf(v, 0.f));
    }
    m = fmaxf(m, __shfl_xor(m, 16));   /* combine rows 0-7 with 8-15 */
    if (lane < 16) out[(size_t)mynode * OUT_DIM + c] = m;
  }
}

extern "C" void kernel_launch(void* const* d_in, const int* in_sizes, int n_in,
                              void* d_out, int out_size, void* d_ws, size_t ws_size,
                              hipStream_t stream) {
  const float* node  = (const float*)d_in[0];
  const float* feat  = (const float*)d_in[1];
  const float* W     = (const float*)d_in[2];
  const float* gamma = (const float*)d_in[3];
  const float* beta  = (const float*)d_in[4];
  const int*   edges = (const int*)d_in[5];
  float* ws  = (float*)d_ws;
  float* out = (float*)d_out;
  (void)in_sizes; (void)n_in; (void)out_size; (void)ws_size;

  kA_stats<<<GRID_A, 256, 0, stream>>>(node, feat, W, edges, ws);
  kB_finalize<<<1, 64, 0, stream>>>(gamma, beta, ws);
  kC_out<<<N_NODES / 8, 256, 0, stream>>>(node, feat, W, edges, ws, out);
}
